// TriplaneAttention_15144054686223
// MI455X (gfx1250) — matrix operand order repaired and checked by reading, not compiled, not run on hardware
//
#include <hip/hip_runtime.h>

typedef _Float16 v16h __attribute__((ext_vector_type(16)));
typedef _Float16 v8h  __attribute__((ext_vector_type(8)));
typedef _Float16 v4h  __attribute__((ext_vector_type(4)));
typedef float    v8f  __attribute__((ext_vector_type(8)));
typedef float    v4f  __attribute__((ext_vector_type(4)));

#define L_SEQ   8192
#define BATCH   4
#define CDIM    256
#define NV      8
#define NHEAD   4
#define DHEAD   64
#define TOK_WG  8
#define ROWS    (TOK_WG * NV)     // 64 context rows per workgroup
#define CP      264               // padded LDS row stride (halves) for ctx/k/v
#define QP      260               // padded LDS row stride (floats) for q

// ---------------------------------------------------------------------------
// Prep: convert Wk/Wv (f32 [256 x 256], row-major [k][n]) into f16 WMMA
// B-fragments in d_ws. Fragment (mat, ntile, kblk) = 512 halves:
//   half[lane*16 + i] = W[kblk*32 + (lane>=16 ? 16 : 0) + i][ntile*16 + (lane&15)]
// i.e. per lane: 16 consecutive K values of one output column (contiguous k
// order, matching the A fragments fed to V_WMMA_F32_16X16X32_F16).
// ---------------------------------------------------------------------------
__global__ __launch_bounds__(256) void prep_weights_f16(
    const float* __restrict__ Wk, const float* __restrict__ Wv,
    _Float16* __restrict__ wsH)
{
  int hi   = blockIdx.x * 256 + threadIdx.x;   // 0 .. 131071
  int i    = hi & 15;
  int lane = (hi >> 4) & 31;
  int kblk = (hi >> 9) & 7;
  int nt   = (hi >> 12) & 15;
  int mat  = hi >> 16;                          // 0 = Wk, 1 = Wv
  int k    = kblk * 32 + ((lane >> 4) << 4) + i;
  int n    = nt * 16 + (lane & 15);
  const float* W = mat ? Wv : Wk;
  wsH[hi] = (_Float16)W[k * CDIM + n];
}

// ---------------------------------------------------------------------------
// Fused: LN(q) + k/v projection (WMMA f16) + per-token 8-way attention.
// 1 block = 256 threads = 8 waves = 8 tokens (64 context rows).
// GEMM loop is K-outer: each B fragment is loaded from L2 exactly once per
// wave; 16 independent accumulators stay live across the K loop.
// ---------------------------------------------------------------------------
__global__ __launch_bounds__(256) void triplane_fused(
    const float* __restrict__ query, const float* __restrict__ context,
    const _Float16* __restrict__ wsH,
    const float* __restrict__ bk, const float* __restrict__ bv,
    const float* __restrict__ gamma, const float* __restrict__ beta,
    float* __restrict__ out, float* __restrict__ probs)
{
  __shared__ _Float16 sCtx[ROWS * CP];
  __shared__ _Float16 sK[ROWS * CP];
  __shared__ _Float16 sV[ROWS * CP];
  __shared__ float    sQ[TOK_WG * QP];
  __shared__ float    sP[TOK_WG * 32];

  const int tid  = threadIdx.x;
  const int wave = tid >> 5;
  const int lane = tid & 31;
  const int blk  = blockIdx.x;                       // 0 .. 4095
  const int b    = blk / (L_SEQ / TOK_WG);
  const int l0   = (blk % (L_SEQ / TOK_WG)) * TOK_WG;

  // ---- Stage context block -> LDS (f32 -> f16), 64 rows x 256 ----
  {
    const float* cbase = context + (size_t)((b * L_SEQ + l0) * NV) * CDIM;
    #pragma unroll
    for (int it = 0; it < 16; ++it) {
      int idx4 = it * 256 + tid;                     // float4 index, 0..4095
      int row  = idx4 >> 6;                          // 64 float4 per row
      int c4   = idx4 & 63;
      v4f d = *(const v4f*)(cbase + (size_t)idx4 * 4);
      v4h h;
      h[0] = (_Float16)d[0]; h[1] = (_Float16)d[1];
      h[2] = (_Float16)d[2]; h[3] = (_Float16)d[3];
      *(v4h*)&sCtx[row * CP + c4 * 4] = h;
    }
  }

  // ---- LayerNorm of this wave's token -> sQ ----
  {
    const int tok = wave;
    const float* qrow = query + (size_t)(b * L_SEQ + l0 + tok) * CDIM;
    v4f q0 = *(const v4f*)(qrow + lane * 8);
    v4f q1 = *(const v4f*)(qrow + lane * 8 + 4);
    float s = 0.f, ss = 0.f;
    #pragma unroll
    for (int j = 0; j < 4; ++j) {
      s  += q0[j] + q1[j];
      ss += q0[j] * q0[j] + q1[j] * q1[j];
    }
    #pragma unroll
    for (int m = 1; m < 32; m <<= 1) {
      s  += __shfl_xor(s,  m, 32);
      ss += __shfl_xor(ss, m, 32);
    }
    float mean = s * (1.f / 256.f);
    float var  = ss * (1.f / 256.f) - mean * mean;
    float rs   = rsqrtf(var + 1e-5f);
    v4f g0 = *(const v4f*)(gamma + lane * 8);
    v4f g1 = *(const v4f*)(gamma + lane * 8 + 4);
    v4f e0 = *(const v4f*)(beta + lane * 8);
    v4f e1 = *(const v4f*)(beta + lane * 8 + 4);
    v4f o0, o1;
    #pragma unroll
    for (int j = 0; j < 4; ++j) {
      o0[j] = (q0[j] - mean) * rs * g0[j] + e0[j];
      o1[j] = (q1[j] - mean) * rs * g1[j] + e1[j];
    }
    *(v4f*)&sQ[tok * QP + lane * 8]     = o0;
    *(v4f*)&sQ[tok * QP + lane * 8 + 4] = o1;
  }
  __syncthreads();

  // ---- GEMM phase: this wave owns N-tiles {2w, 2w+1} for K and V ----
  {
    const int nt0  = wave * 2, nt1 = wave * 2 + 1;
    const int col  = lane & 15;
    const int rsel = lane >> 4;                       // 0 or 1
    const float bk0 = bk[nt0 * 16 + col], bk1 = bk[nt1 * 16 + col];
    const float bv0 = bv[nt0 * 16 + col], bv1 = bv[nt1 * 16 + col];

    v8f aK0[4], aK1[4], aV0[4], aV1[4];
    #pragma unroll
    for (int mt = 0; mt < 4; ++mt) {
      aK0[mt] = (v8f){}; aK1[mt] = (v8f){};
      aV0[mt] = (v8f){}; aV1[mt] = (v8f){};
    }

    #pragma unroll
    for (int kb = 0; kb < 8; ++kb) {
      // B fragments: pre-swizzled in ws, one contiguous 32B read per lane,
      // loaded exactly once per wave (hits L2; weights are only 256 KB).
      v16h bK0 = *(const v16h*)(wsH + ((size_t)((0*16 + nt0)*8 + kb))*512 + lane*16);
      v16h bK1 = *(const v16h*)(wsH + ((size_t)((0*16 + nt1)*8 + kb))*512 + lane*16);
      v16h bV0 = *(const v16h*)(wsH + ((size_t)((1*16 + nt0)*8 + kb))*512 + lane*16);
      v16h bV1 = *(const v16h*)(wsH + ((size_t)((1*16 + nt1)*8 + kb))*512 + lane*16);
      const int kbase = kb * 32 + (rsel << 4);
      #pragma unroll
      for (int mt = 0; mt < 4; ++mt) {
        // A fragment (contiguous k order, as in B): lane<16 -> K{0..15}; lane>=16 -> K{16..31}
        const int arow = mt * 16 + (lane & 15);
        v8h alo = *(const v8h*)&sCtx[arow * CP + kbase];
        v8h ahi = *(const v8h*)&sCtx[arow * CP + kbase + 8];
        v16h a  = __builtin_shufflevector(alo, ahi,
                    0,1,2,3,4,5,6,7,8,9,10,11,12,13,14,15);
        aK0[mt] = __builtin_amdgcn_wmma_f32_16x16x32_f16(false, a, false, bK0, (short)0, aK0[mt], false, false);
        aK1[mt] = __builtin_amdgcn_wmma_f32_16x16x32_f16(false, a, false, bK1, (short)0, aK1[mt], false, false);
        aV0[mt] = __builtin_amdgcn_wmma_f32_16x16x32_f16(false, a, false, bV0, (short)0, aV0[mt], false, false);
        aV1[mt] = __builtin_amdgcn_wmma_f32_16x16x32_f16(false, a, false, bV1, (short)0, aV1[mt], false, false);
      }
    }

    // epilogue: + bias, f32 -> f16 into sK/sV (C/D layout: acc[r] -> row r + 8*rsel)
    #pragma unroll
    for (int mt = 0; mt < 4; ++mt) {
      const int rowbase = mt * 16 + rsel * 8;
      #pragma unroll
      for (int r = 0; r < 8; ++r) {
        const int row = rowbase + r;
        sK[row * CP + nt0 * 16 + col] = (_Float16)(aK0[mt][r] + bk0);
        sK[row * CP + nt1 * 16 + col] = (_Float16)(aK1[mt][r] + bk1);
        sV[row * CP + nt0 * 16 + col] = (_Float16)(aV0[mt][r] + bv0);
        sV[row * CP + nt1 * 16 + col] = (_Float16)(aV1[mt][r] + bv1);
      }
    }
  }
  __syncthreads();

  // ---- Attention: one wave per token; lane = (head h = lane>>3, n = lane&7) ----
  {
    const int tok = wave;
    const int h = lane >> 3, n = lane & 7;
    const _Float16* krow = &sK[(tok * NV + n) * CP + h * DHEAD];
    const float*    qh   = &sQ[tok * QP + h * DHEAD];
    float sc = 0.f;
    #pragma unroll
    for (int dd = 0; dd < 8; ++dd) {
      v8h kv = *(const v8h*)(krow + dd * 8);
      v4f qa = *(const v4f*)(qh + dd * 8);
      v4f qb = *(const v4f*)(qh + dd * 8 + 4);
      sc += qa[0]*(float)kv[0] + qa[1]*(float)kv[1] + qa[2]*(float)kv[2] + qa[3]*(float)kv[3]
          + qb[0]*(float)kv[4] + qb[1]*(float)kv[5] + qb[2]*(float)kv[6] + qb[3]*(float)kv[7];
    }
    sc *= 0.125f;                                     // 1/sqrt(D_HEAD)
    float mx = sc;
    #pragma unroll
    for (int m = 1; m < 8; m <<= 1) mx = fmaxf(mx, __shfl_xor(mx, m, 32));
    float e = __expf(sc - mx);
    float sum = e;
    #pragma unroll
    for (int m = 1; m < 8; m <<= 1) sum += __shfl_xor(sum, m, 32);
    float p = e / sum;
    probs[((size_t)(b * NHEAD + h) * L_SEQ + (l0 + tok)) * NV + n] = p;
    sP[tok * 32 + lane] = p;
  }
  __syncthreads();

  // ---- Combine: out[c] = sum_n p[h(c),n] * v[n][c]; lane covers 8 channels ----
  {
    const int tok = wave;
    const int h  = lane >> 3;                         // 8 channels/lane, 64/head
    const int c0 = lane * 8;
    float acc[8] = {0,0,0,0,0,0,0,0};
    #pragma unroll
    for (int n = 0; n < 8; ++n) {
      float p = sP[tok * 32 + h * 8 + n];
      v8h vv = *(const v8h*)&sV[(tok * NV + n) * CP + c0];
      #pragma unroll
      for (int j = 0; j < 8; ++j) acc[j] += p * (float)vv[j];
    }
    float* orow = out + (size_t)(b * L_SEQ + l0 + tok) * CDIM + c0;
    v4f o0, o1;
    #pragma unroll
    for (int j = 0; j < 4; ++j) { o0[j] = acc[j]; o1[j] = acc[4 + j]; }
    *(v4f*)orow       = o0;
    *(v4f*)(orow + 4) = o1;
  }
}

extern "C" void kernel_launch(void* const* d_in, const int* in_sizes, int n_in,
                              void* d_out, int out_size, void* d_ws, size_t ws_size,
                              hipStream_t stream) {
  (void)in_sizes; (void)n_in; (void)out_size; (void)ws_size;
  const float* query   = (const float*)d_in[0];
  const float* context = (const float*)d_in[1];
  const float* Wk      = (const float*)d_in[2];
  const float* bk      = (const float*)d_in[3];
  const float* Wv      = (const float*)d_in[4];
  const float* bv      = (const float*)d_in[5];
  const float* gamma   = (const float*)d_in[6];
  const float* beta    = (const float*)d_in[7];
  float* out   = (float*)d_out;
  float* probs = out + (size_t)BATCH * L_SEQ * CDIM;  // outputs concatenated
  _Float16* wsH = (_Float16*)d_ws;                    // needs 256 KB

  // f32 -> f16 weight fragment swizzle (2 * 256*256 halves, one per thread)
  prep_weights_f16<<<512, 256, 0, stream>>>(Wk, Wv, wsH);
  // fused LN + projection + attention: 4096 blocks of 8 waves
  triplane_fused<<<(BATCH * L_SEQ) / TOK_WG, 256, 0, stream>>>(
      query, context, wsH, bk, bv, gamma, beta, out, probs);
}
